// BailingMoEDecoderLayer_24266565222514
// MI455X (gfx1250) — matrix operand order repaired and checked by reading, not compiled, not run on hardware
//
#include <hip/hip_runtime.h>

typedef __attribute__((ext_vector_type(16))) __bf16 v16bf;
typedef __attribute__((ext_vector_type(2)))  __bf16 v2bf;
typedef __attribute__((ext_vector_type(8)))  float  v8f;
typedef __attribute__((ext_vector_type(2)))  float  v2f;
typedef __attribute__((ext_vector_type(4)))  unsigned int v4u;
typedef __attribute__((ext_vector_type(8)))  int v8i;

#define T_TOK 2048
#define D_DIM 2048
#define HQ_N  16
#define HKV_N 4
#define HD_N  128
#define S_LEN 1024
#define B_N   2
#define E_N   32
#define K_TOP 4
#define F_DIM 1024
#define FS_DIM 2048

__device__ __forceinline__ unsigned pack_bf16(float x, float y) {
  v2f f = {x, y};
  v2bf h = __builtin_convertvector(f, v2bf);
  union { v2bf h; unsigned u; } t; t.h = h; return t.u;
}

__device__ __forceinline__ float silu_f(float x) { return x / (1.f + __expf(-x)); }

// DPP16 butterfly reduce within each 16-lane half (rows of wave32).
#define DPPF(x, ctrl) __builtin_bit_cast(float, \
    __builtin_amdgcn_update_dpp(0, __builtin_bit_cast(int, (x)), (ctrl), 0xF, 0xF, true))
__device__ __forceinline__ float red16_max(float x) {
  x = fmaxf(x, DPPF(x, 0xB1));   // quad_perm [1,0,3,2]
  x = fmaxf(x, DPPF(x, 0x4E));   // quad_perm [2,3,0,1]
  x = fmaxf(x, DPPF(x, 0x141));  // row_half_mirror
  x = fmaxf(x, DPPF(x, 0x140));  // row_mirror
  return x;
}
__device__ __forceinline__ float red16_sum(float x) {
  x += DPPF(x, 0xB1); x += DPPF(x, 0x4E); x += DPPF(x, 0x141); x += DPPF(x, 0x140);
  return x;
}

// LDS byte offset of a __shared__ object (generic -> AS3 -> int).
#define LDS_OFF(p) ((unsigned)(unsigned long long)(__attribute__((address_space(3))) const void*)(p))

// 2D TDM load: tile (32 x rows) fp32, row stride Kd floats, LDS row stride 34 dw (pad 2).
__device__ __forceinline__ void tdm_load_a_tile(const float* gptr, unsigned lds_addr,
                                                int Kd, int rows) {
  union { unsigned w[4]; v4u v; } g0;
  union { unsigned long long q[4]; v8i v; } g1;
  unsigned long long ga = (unsigned long long)gptr;
  g0.w[0] = 1u;
  g0.w[1] = lds_addr;
  g0.w[2] = (unsigned)(ga & 0xffffffffull);
  g0.w[3] = (unsigned)((ga >> 32) & 0x01ffffffull) | (2u << 30);
  g1.q[0] = (2ull << 16) | (1ull << 20) | (4ull << 22) | (1ull << 25)
          | ((unsigned long long)(Kd & 0xffff) << 48);
  g1.q[1] = (unsigned long long)((unsigned)Kd >> 16)
          | (0x100000ull << 16)
          | (32ull << 48);
  g1.q[2] = (unsigned long long)(unsigned)rows
          | ((unsigned long long)(unsigned)Kd << 32);
  g1.q[3] = 0ull;
  union { unsigned long long q[2]; v4u v; } z2;
  z2.q[0] = z2.q[1] = 0ull;
  asm volatile("tensor_load_to_lds %0, %1, %2, %3"
               :: "s"(g0.v), "s"(g1.v), "s"(z2.v), "s"(z2.v)
               : "memory");
}

// ---------------------------------------------------------------------------
__global__ __launch_bounds__(256) void rmsnorm_kernel(
    const float* __restrict__ x, const float* __restrict__ sc,
    float* __restrict__ y, int D)
{
  long row = blockIdx.x;
  const float* xr = x + row * D;
  float ss = 0.f;
  for (int d = threadIdx.x; d < D; d += 256) { float v = xr[d]; ss += v * v; }
  #pragma unroll
  for (int off = 16; off >= 1; off >>= 1) ss += __shfl_xor(ss, off);
  __shared__ float wr[8];
  if ((threadIdx.x & 31) == 0) wr[threadIdx.x >> 5] = ss;
  __syncthreads();
  float tot = 0.f;
  #pragma unroll
  for (int i = 0; i < 8; ++i) tot += wr[i];
  float r = rsqrtf(tot / (float)D + 1e-6f);
  for (int d = threadIdx.x; d < 2048; d += 256) y[row * D + d] = xr[d] * r * sc[d];
}

// ---------------------------------------------------------------------------
// bf16 WMMA GEMM, block tile 128(M) x 64(N), k-step 32, 8 waves (4M x 2N),
// 4 WMMAs per wave per k-step.  A tile: fp32 in LDS via double-buffered TDM
// (contiguous-row modes) or manual gather staging.  B tile: register-prefetched
// one k-step ahead, stored transposed bf16.
// MODE 0: C=acc            1: C=acc+R           2: gather A -> slots
// MODE 3: A=slots, scatter atomicAdd(w*acc)     4: dense C=silu(C)*acc
// MODE 5: gather A, slots C=silu(C)*acc
template <int MODE>
__global__ __launch_bounds__(256) void gemm_bf16(
    const float* __restrict__ A, const float* __restrict__ Bm,
    float* __restrict__ C, const float* __restrict__ R,
    const int* __restrict__ lists, const float* __restrict__ wsl,
    const int* __restrict__ counts, const int* __restrict__ offs,
    int M, int N, int Kd, long bstride)
{
  constexpr bool IS_EXPERT = (MODE == 2 || MODE == 3 || MODE == 5);
  constexpr bool GATHER_A  = (MODE == 2 || MODE == 5);

  __shared__ float  Asf[2][128 * 34];
  __shared__ __bf16 Bs[64 * 34];

  int e = blockIdx.z;
  const float* Bp = Bm + (long)e * bstride;
  int gm0 = blockIdx.y * 128, gn0 = blockIdx.x * 64;
  int cnt = M, obase = 0;
  if (IS_EXPERT) { cnt = counts[e]; if (gm0 >= cnt) return; obase = offs[e]; }

  int tid = threadIdx.x, lane = tid & 31, w = tid >> 5;
  int wm = w >> 1, wn = w & 1;
  int lh = lane & 15, hi = lane >> 4;

  v8f acc[2][2];
  #pragma unroll
  for (int i = 0; i < 2; ++i)
    #pragma unroll
    for (int j = 0; j < 2; ++j)
      #pragma unroll
      for (int q = 0; q < 8; ++q) acc[i][j][q] = 0.f;

  long rowbase = (MODE == 3) ? (long)obase + gm0 : (long)gm0;
  const float* Abase = A + rowbase * Kd;
  unsigned a_lds0 = LDS_OFF(&Asf[0][0]);
  unsigned a_lds1 = LDS_OFF(&Asf[1][0]);
  unsigned* Bsu = (unsigned*)&Bs[0];

  float br[8];
  auto load_breg = [&](int kk) {
    #pragma unroll
    for (int i = 0; i < 4; ++i) {
      int idx = tid + i * 256;
      int n = idx & 63, kp = idx >> 6;
      const float* bp = &Bp[(long)(kk + 2 * kp) * N + gn0 + n];
      br[2 * i] = bp[0]; br[2 * i + 1] = bp[N];
    }
  };

  int nk = Kd >> 5;
  load_breg(0);
  if (!GATHER_A && w == 0) tdm_load_a_tile(Abase, a_lds0, Kd, 128);

  for (int i = 0; i < nk; ++i) {
    int kk = i << 5;
    __syncthreads();                         // previous tile fully consumed
    #pragma unroll
    for (int t2 = 0; t2 < 4; ++t2) {         // commit prefetched B regs to LDS
      int idx = tid + t2 * 256;
      int n = idx & 63, kp = idx >> 6;
      Bsu[(n * 34 + 2 * kp) >> 1] = pack_bf16(br[2 * t2], br[2 * t2 + 1]);
    }
    if (GATHER_A) {                          // manual gather staging: 128x32 fp32
      #pragma unroll
      for (int i2 = 0; i2 < 8; ++i2) {
        int idx = tid + i2 * 256;
        int r = idx >> 4, c2 = (idx & 15) * 2;
        int ri = gm0 + r; if (ri >= cnt) ri = cnt - 1;
        long arow = (long)lists[e * T_TOK + ri];
        float2 v = *(const float2*)&A[arow * Kd + kk + c2];
        *(float2*)&Asf[0][r * 34 + c2] = v;
      }
    }
    if (i + 1 < nk) {
      load_breg(kk + 32);                    // global B loads overlap compute
      if (!GATHER_A && w == 0)
        tdm_load_a_tile(Abase + kk + 32, ((i + 1) & 1) ? a_lds1 : a_lds0, Kd, 128);
    }
    if (!GATHER_A && w == 0) {
      if (i + 1 < nk) __builtin_amdgcn_s_wait_tensorcnt(1);   // tile i complete
      else            __builtin_amdgcn_s_wait_tensorcnt(0);
    }
    __syncthreads();

    const float* Acur = GATHER_A ? &Asf[0][0] : &Asf[i & 1][0];
    v16bf a0, a1;
    const float* ar0 = &Acur[(wm * 32 + lh) * 34];
    const float* ar1 = &Acur[(wm * 32 + 16 + lh) * 34];
    #pragma unroll
    for (int q = 0; q < 4; ++q) {
      float2 p0 = *(const float2*)&ar0[2 * q + hi * 8];
      float2 p1 = *(const float2*)&ar0[16 + 2 * q + hi * 8];
      a0[2*q] = (__bf16)p0.x; a0[2*q+1] = (__bf16)p0.y;
      a0[8+2*q] = (__bf16)p1.x; a0[9+2*q] = (__bf16)p1.y;
      float2 q0 = *(const float2*)&ar1[2 * q + hi * 8];
      float2 q1 = *(const float2*)&ar1[16 + 2 * q + hi * 8];
      a1[2*q] = (__bf16)q0.x; a1[2*q+1] = (__bf16)q0.y;
      a1[8+2*q] = (__bf16)q1.x; a1[9+2*q] = (__bf16)q1.y;
    }
    v16bf b0, b1;
    const __bf16* b0r = &Bs[(wn * 32 + lh) * 34 + hi * 8];
    const __bf16* b1r = &Bs[(wn * 32 + 16 + lh) * 34 + hi * 8];
    #pragma unroll
    for (int j = 0; j < 16; ++j) { b0[j] = b0r[(j & 7) + (j >> 3) * 16]; b1[j] = b1r[(j & 7) + (j >> 3) * 16]; }

    acc[0][0] = __builtin_amdgcn_wmma_f32_16x16x32_bf16(false, a0, false, b0, (short)0, acc[0][0], false, false);
    acc[0][1] = __builtin_amdgcn_wmma_f32_16x16x32_bf16(false, a0, false, b1, (short)0, acc[0][1], false, false);
    acc[1][0] = __builtin_amdgcn_wmma_f32_16x16x32_bf16(false, a1, false, b0, (short)0, acc[1][0], false, false);
    acc[1][1] = __builtin_amdgcn_wmma_f32_16x16x32_bf16(false, a1, false, b1, (short)0, acc[1][1], false, false);
  }

  #pragma unroll
  for (int am = 0; am < 2; ++am) {
    #pragma unroll
    for (int j = 0; j < 8; ++j) {
      int r = gm0 + wm * 32 + am * 16 + j + 8 * hi;
      int c0 = gn0 + wn * 32 + lh, c1 = c0 + 16;
      if (MODE == 0) {
        C[(long)r * N + c0] = acc[am][0][j]; C[(long)r * N + c1] = acc[am][1][j];
      } else if (MODE == 1) {
        C[(long)r * N + c0] = acc[am][0][j] + R[(long)r * N + c0];
        C[(long)r * N + c1] = acc[am][1][j] + R[(long)r * N + c1];
      } else if (MODE == 2) {
        if (r < cnt) {
          long orow = obase + r;
          C[orow * N + c0] = acc[am][0][j]; C[orow * N + c1] = acc[am][1][j];
        }
      } else if (MODE == 4) {
        float g0 = C[(long)r * N + c0], g1 = C[(long)r * N + c1];
        C[(long)r * N + c0] = silu_f(g0) * acc[am][0][j];
        C[(long)r * N + c1] = silu_f(g1) * acc[am][1][j];
      } else if (MODE == 5) {
        if (r < cnt) {
          long orow = obase + r;
          float g0 = C[orow * N + c0], g1 = C[orow * N + c1];
          C[orow * N + c0] = silu_f(g0) * acc[am][0][j];
          C[orow * N + c1] = silu_f(g1) * acc[am][1][j];
        }
      } else {
        if (r < cnt) {
          int tkn = lists[e * T_TOK + r]; float wgt = wsl[e * T_TOK + r];
          atomicAdd(&C[(long)tkn * N + c0], wgt * acc[am][0][j]);
          atomicAdd(&C[(long)tkn * N + c1], wgt * acc[am][1][j]);
        }
      }
    }
  }
}

// ---------------------------------------------------------------------------
__global__ __launch_bounds__(64) void qknorm_rope_kernel(
    const int* __restrict__ positions, float* __restrict__ qbuf, float* __restrict__ kbuf,
    const float* __restrict__ qn, const float* __restrict__ kn)
{
  int t = blockIdx.x, h = blockIdx.y, i = threadIdx.x;
  float* base = (h < HQ_N) ? (qbuf + (long)t * (HQ_N * HD_N) + h * HD_N)
                           : (kbuf + (long)t * (HKV_N * HD_N) + (h - HQ_N) * HD_N);
  const float* sc = (h < HQ_N) ? qn : kn;
  float x1 = base[i], x2 = base[i + 64];
  float ss = x1 * x1 + x2 * x2;
  #pragma unroll
  for (int off = 16; off >= 1; off >>= 1) ss += __shfl_xor(ss, off);
  __shared__ float w2[2];
  if ((threadIdx.x & 31) == 0) w2[threadIdx.x >> 5] = ss;
  __syncthreads();
  float r = rsqrtf((w2[0] + w2[1]) / 128.f + 1e-6f);
  x1 = x1 * r * sc[i];
  x2 = x2 * r * sc[i + 64];
  float pos = (float)positions[t];
  float inv = __expf(-((float)(2 * i) / 128.f) * logf(1.0e6f));
  float ang = pos * inv, c = __cosf(ang), s = __sinf(ang);
  base[i]      = x1 * c - x2 * s;
  base[i + 64] = x1 * s + x2 * c;
}

// ---------------------------------------------------------------------------
__global__ __launch_bounds__(256) void flash_attn_kernel(
    const float* __restrict__ qbuf, const float* __restrict__ kbuf,
    const float* __restrict__ vbuf, float* __restrict__ attn)
{
  __shared__ __bf16 Kt[32][136];
  __shared__ __bf16 Vt[32][136];
  __shared__ __bf16 Pt[8][16][34];

  int lane = threadIdx.x & 31, w = threadIdx.x >> 5;
  int lh = lane & 15, hi = lane >> 4;
  int qb = blockIdx.x, hq = blockIdx.y, b = blockIdx.z;
  int hkv = hq >> 2;
  int q0 = qb * 128;
  int qrow = q0 + w * 16 + lh;

  const float* qptr = qbuf + (long)(b * S_LEN + qrow) * (HQ_N * HD_N) + hq * HD_N;
  v16bf qa[4];
  #pragma unroll
  for (int f = 0; f < 4; ++f) {
    int kc = f * 32;
    #pragma unroll
    for (int i = 0; i < 4; ++i) {
      float2 p0 = *(const float2*)&qptr[kc + 2*i + hi*8];
      float2 p1 = *(const float2*)&qptr[kc + 16 + 2*i + hi*8];
      qa[f][2*i] = (__bf16)p0.x; qa[f][2*i+1] = (__bf16)p0.y;
      qa[f][8+2*i] = (__bf16)p1.x; qa[f][9+2*i] = (__bf16)p1.y;
    }
  }

  float m[8], l[8];
  v8f o[8];
  #pragma unroll
  for (int j = 0; j < 8; ++j) { m[j] = -1e30f; l[j] = 0.f; }
  #pragma unroll
  for (int t = 0; t < 8; ++t)
    #pragma unroll
    for (int j = 0; j < 8; ++j) o[t][j] = 0.f;

  const float scale = 0.08838834764831845f;   // 1/sqrt(128)
  unsigned* Ktu = (unsigned*)&Kt[0][0];
  unsigned* Vtu = (unsigned*)&Vt[0][0];

  for (int kc0 = 0; kc0 < q0 + 128; kc0 += 32) {
    __syncthreads();
    #pragma unroll
    for (int i = 0; i < 8; ++i) {
      int idx = threadIdx.x + i * 256;
      int r = idx >> 6, c2 = (idx & 63) * 2;
      long gro = (long)(b * S_LEN + kc0 + r) * (HKV_N * HD_N) + hkv * HD_N + c2;
      float2 kv = *(const float2*)&kbuf[gro];
      float2 vv = *(const float2*)&vbuf[gro];
      Ktu[(r * 136 + c2) >> 1] = pack_bf16(kv.x, kv.y);
      Vtu[(r * 136 + c2) >> 1] = pack_bf16(vv.x, vv.y);
    }
    __syncthreads();

    v8f s0, s1;
    #pragma unroll
    for (int q = 0; q < 8; ++q) { s0[q] = 0.f; s1[q] = 0.f; }
    #pragma unroll
    for (int f = 0; f < 4; ++f) {
      v16bf b0, b1;
      #pragma unroll
      for (int j = 0; j < 16; ++j) {
        b0[j] = Kt[lh][f*32 + hi*8 + (j&7) + (j>>3)*16];
        b1[j] = Kt[16 + lh][f*32 + hi*8 + (j&7) + (j>>3)*16];
      }
      s0 = __builtin_amdgcn_wmma_f32_16x16x32_bf16(false, qa[f], false, b0, (short)0, s0, false, false);
      s1 = __builtin_amdgcn_wmma_f32_16x16x32_bf16(false, qa[f], false, b1, (short)0, s1, false, false);
    }

    #pragma unroll
    for (int j = 0; j < 8; ++j) {
      int r = q0 + w * 16 + j + 8 * hi;
      float v0 = s0[j] * scale; if (kc0 + lh > r)      v0 = -1e30f;
      float v1 = s1[j] * scale; if (kc0 + 16 + lh > r) v1 = -1e30f;
      float mn = fmaxf(m[j], red16_max(fmaxf(v0, v1)));
      float a  = __expf(m[j] - mn);
      float p0 = __expf(v0 - mn), p1 = __expf(v1 - mn);
      float rs = red16_sum(p0 + p1);
      l[j] = l[j] * a + rs;
      m[j] = mn;
      #pragma unroll
      for (int t = 0; t < 8; ++t) o[t][j] *= a;
      Pt[w][j + 8*hi][lh]      = (__bf16)p0;
      Pt[w][j + 8*hi][16 + lh] = (__bf16)p1;
    }

    v16bf pa;
    #pragma unroll
    for (int i = 0; i < 4; ++i) { int k0 = 2*i + hi*8;      pa[2*i] = Pt[w][lh][k0]; pa[2*i+1] = Pt[w][lh][k0+1]; }
    #pragma unroll
    for (int i = 0; i < 4; ++i) { int k0 = 16 + 2*i + hi*8; pa[8+2*i] = Pt[w][lh][k0]; pa[9+2*i] = Pt[w][lh][k0+1]; }

    #pragma unroll
    for (int t = 0; t < 8; ++t) {
      v16bf vb;
      #pragma unroll
      for (int j = 0; j < 16; ++j) vb[j] = Vt[hi*8 + (j&7) + (j>>3)*16][t*16 + lh];
      o[t] = __builtin_amdgcn_wmma_f32_16x16x32_bf16(false, pa, false, vb, (short)0, o[t], false, false);
    }
  }

  #pragma unroll
  for (int j = 0; j < 8; ++j) {
    int r = q0 + w * 16 + j + 8 * hi;
    float li = 1.f / l[j];
    #pragma unroll
    for (int t = 0; t < 8; ++t)
      attn[(long)(b * S_LEN + r) * (HQ_N * HD_N) + hq * HD_N + t*16 + lh] = o[t][j] * li;
  }
}

// ---------------------------------------------------------------------------
__global__ void zero_counts_kernel(int* c) { if (threadIdx.x < E_N) c[threadIdx.x] = 0; }

__global__ void prefix_kernel(const int* __restrict__ counts, int* __restrict__ offs)
{
  if (threadIdx.x == 0) { int s = 0; for (int e = 0; e < E_N; ++e) { offs[e] = s; s += counts[e]; } offs[E_N] = s; }
}

__global__ __launch_bounds__(256) void router_kernel(
    const float* __restrict__ x, const float* __restrict__ gw,
    int* __restrict__ counts, int* __restrict__ lists, float* __restrict__ wsl)
{
  __shared__ float tv[8][4];
  __shared__ int   ti[8][4];
  int lane = threadIdx.x & 31, w = threadIdx.x >> 5;
  int t = blockIdx.x * 8 + w;
  const float* xr = x + (long)t * D_DIM;
  float acc = 0.f;
  for (int d = 0; d < D_DIM; ++d) acc += xr[d] * gw[d * E_N + lane];
  float g = 1.f / (1.f + __expf(-acc));
  for (int kk = 0; kk < K_TOP; ++kk) {
    float v = g; int idx = lane;
    #pragma unroll
    for (int off = 16; off >= 1; off >>= 1) {
      float ov = __shfl_xor(v, off);
      int   oi = __shfl_xor(idx, off);
      if (ov > v || (ov == v && oi < idx)) { v = ov; idx = oi; }
    }
    if (lane == 0) { tv[w][kk] = v; ti[w][kk] = idx; }
    if (lane == idx) g = -1e30f;
  }
  if (lane < K_TOP) {
    float wsum = tv[w][0] + tv[w][1] + tv[w][2] + tv[w][3];
    int e = ti[w][lane];
    int pos = atomicAdd(&counts[e], 1);
    lists[e * T_TOK + pos] = t;
    wsl[e * T_TOK + pos]   = tv[w][lane] / wsum;
  }
}

// ---------------------------------------------------------------------------
extern "C" void kernel_launch(void* const* d_in, const int* in_sizes, int n_in,
                              void* d_out, int out_size, void* d_ws, size_t ws_size,
                              hipStream_t stream)
{
  const int*   positions = (const int*)  d_in[0];
  const float* hidden    = (const float*)d_in[1];
  const float* in_ln     = (const float*)d_in[2];
  const float* post_ln   = (const float*)d_in[3];
  const float* q_norm    = (const float*)d_in[4];
  const float* k_norm    = (const float*)d_in[5];
  const float* wq        = (const float*)d_in[6];
  const float* wk        = (const float*)d_in[7];
  const float* wv        = (const float*)d_in[8];
  const float* wo        = (const float*)d_in[9];
  const float* gate_w    = (const float*)d_in[10];
  const float* sh_wg     = (const float*)d_in[11];
  const float* sh_wu     = (const float*)d_in[12];
  const float* sh_wd     = (const float*)d_in[13];
  const float* e_wg      = (const float*)d_in[14];
  const float* e_wu      = (const float*)d_in[15];
  const float* e_wd      = (const float*)d_in[16];
  float* out = (float*)d_out;

  float* f       = (float*)d_ws;
  float* h_norm  = f;
  float* qbuf    = h_norm + (long)T_TOK * D_DIM;
  float* kbuf    = qbuf  + (long)T_TOK * (HQ_N * HD_N);
  float* vbuf    = kbuf  + (long)T_TOK * (HKV_N * HD_N);
  float* attn    = vbuf  + (long)T_TOK * (HKV_N * HD_N);
  float* x2      = attn  + (long)T_TOK * (HQ_N * HD_N);
  float* h_post  = x2    + (long)T_TOK * D_DIM;
  float* shg     = h_post+ (long)T_TOK * D_DIM;
  float* slots_g = shg   + (long)T_TOK * FS_DIM;
  float* wsl     = slots_g + (long)(T_TOK * K_TOP) * F_DIM;     // E*T
  int*   counts  = (int*)(wsl + (long)E_N * T_TOK);
  int*   offs    = counts + E_N;
  int*   lists   = offs + E_N + 2;

  rmsnorm_kernel<<<T_TOK, 256, 0, stream>>>(hidden, in_ln, h_norm, D_DIM);

  gemm_bf16<0><<<dim3(32, 16, 1), 256, 0, stream>>>(h_norm, wq, qbuf, nullptr, nullptr, nullptr, nullptr, nullptr, T_TOK, HQ_N*HD_N, D_DIM, 0);
  gemm_bf16<0><<<dim3(8,  16, 1), 256, 0, stream>>>(h_norm, wk, kbuf, nullptr, nullptr, nullptr, nullptr, nullptr, T_TOK, HKV_N*HD_N, D_DIM, 0);
  gemm_bf16<0><<<dim3(8,  16, 1), 256, 0, stream>>>(h_norm, wv, vbuf, nullptr, nullptr, nullptr, nullptr, nullptr, T_TOK, HKV_N*HD_N, D_DIM, 0);

  qknorm_rope_kernel<<<dim3(T_TOK, HQ_N + HKV_N), 64, 0, stream>>>(positions, qbuf, kbuf, q_norm, k_norm);

  flash_attn_kernel<<<dim3(S_LEN/128, HQ_N, B_N), 256, 0, stream>>>(qbuf, kbuf, vbuf, attn);

  gemm_bf16<1><<<dim3(32, 16, 1), 256, 0, stream>>>(attn, wo, x2, hidden, nullptr, nullptr, nullptr, nullptr, T_TOK, D_DIM, HQ_N*HD_N, 0);

  rmsnorm_kernel<<<T_TOK, 256, 0, stream>>>(x2, post_ln, h_post, D_DIM);

  // shared SwiGLU: gate -> shg; up fused with silu*mul into shg; down + residual -> out
  gemm_bf16<0><<<dim3(32, 16, 1), 256, 0, stream>>>(h_post, sh_wg, shg, nullptr, nullptr, nullptr, nullptr, nullptr, T_TOK, FS_DIM, D_DIM, 0);
  gemm_bf16<4><<<dim3(32, 16, 1), 256, 0, stream>>>(h_post, sh_wu, shg, nullptr, nullptr, nullptr, nullptr, nullptr, T_TOK, FS_DIM, D_DIM, 0);
  gemm_bf16<1><<<dim3(32, 16, 1), 256, 0, stream>>>(shg, sh_wd, out, x2, nullptr, nullptr, nullptr, nullptr, T_TOK, D_DIM, FS_DIM, 0);

  zero_counts_kernel<<<1, 32, 0, stream>>>(counts);
  router_kernel<<<T_TOK/8, 256, 0, stream>>>(h_post, gate_w, counts, lists, wsl);
  prefix_kernel<<<1, 32, 0, stream>>>(counts, offs);

  // routed experts: gate -> slots; up fused silu*mul -> slots; weighted scatter down -> out
  gemm_bf16<2><<<dim3(16, 16, E_N), 256, 0, stream>>>(h_post, e_wg, slots_g, nullptr, lists, wsl, counts, offs, T_TOK, F_DIM, D_DIM, (long)D_DIM * F_DIM);
  gemm_bf16<5><<<dim3(16, 16, E_N), 256, 0, stream>>>(h_post, e_wu, slots_g, nullptr, lists, wsl, counts, offs, T_TOK, F_DIM, D_DIM, (long)D_DIM * F_DIM);
  gemm_bf16<3><<<dim3(32, 16, E_N), 256, 0, stream>>>(slots_g, e_wd, out, nullptr, lists, wsl, counts, offs, T_TOK, D_DIM, F_DIM, (long)F_DIM * D_DIM);
}
